// SpikingCPG_48464410968270
// MI455X (gfx1250) — matrix operand order repaired and checked by reading, not compiled, not run on hardware
//
#include <hip/hip_runtime.h>

// Spiking CPG: B=2048 x N=80 x T=400. Bandwidth floor ~545MB (~23us @23.3TB/s).
// sf@W.T (10.5 GFLOP) runs on v_wmma_f32_16x16x32_f16 (3/step/wave, K padded to 96).
// One workgroup = 5 waves = one 16-batch tile; wave w owns output neurons
// [16w,16w+16). State kept in WMMA C-layout registers.
// Round-2 latency fixes:
//   * inline-asm split barrier (s_wait_dscnt only) -> no per-step global
//     store-ack drain that __syncthreads() was forcing,
//   * double-buffered spike LDS -> 1 barrier/step instead of 2,
//   * software-pipelined noise loads (t+1 issued before barrier t) + L2
//     prefetch of t+2.

#define N_  80
#define B_  2048
#define T_  400
#define STR 104   // LDS row stride in halves: 208 B = 13*16, keeps b128 aligned
#define DT_ 0.1f

typedef __attribute__((ext_vector_type(16))) _Float16 v16h;
typedef __attribute__((ext_vector_type(8)))  float    v8f;

union AU { v16h h; uint4 q[2]; };
union BU { v16h h; unsigned u[8]; };

__device__ __forceinline__ unsigned pack_h2(float a, float b) {
  _Float16 ha = (_Float16)a, hb = (_Float16)b;
  unsigned short ua = __builtin_bit_cast(unsigned short, ha);
  unsigned short ub = __builtin_bit_cast(unsigned short, hb);
  return (unsigned)ua | ((unsigned)ub << 16);
}

// Workgroup barrier that drains ONLY the LDS counter (not global store acks).
__device__ __forceinline__ void lds_barrier() {
  asm volatile(
      "s_wait_dscnt 0x0\n\t"
      "s_barrier_signal -1\n\t"
      "s_barrier_wait -1" ::: "memory");
}

__global__ __launch_bounds__(160) void SpikingCPG_kernel(
    const float* __restrict__ v0,   const float* __restrict__ rf0,
    const float* __restrict__ is0,  const float* __restrict__ ad0,
    const float* __restrict__ noise,const float* __restrict__ dmod,
    const float* __restrict__ W,    const float* __restrict__ tau_m,
    const float* __restrict__ vth,  const float* __restrict__ drive,
    const float* __restrict__ rstv, float* __restrict__ out_v,
    float* __restrict__ out_motor)
{
  const int tid  = threadIdx.x;
  const int wv   = tid >> 5;       // 0..4 : output neuron tile
  const int lane = tid & 31;
  const int col  = lane & 15;      // neuron column within tile (C-layout)
  const int hi   = lane >> 4;      // 0: batch rows 0-7, 1: rows 8-15
  const int b0   = blockIdx.x * 16;
  const int n    = wv * 16 + col;  // this lane's global neuron index

  __shared__ _Float16 spk[2 * 16 * STR];   // double-buffered spike tile
  for (int i = tid; i < 2 * 16 * STR; i += 160) spk[i] = (_Float16)0.f;

  const float dm     = dmod[0];
  const float drv    = drive[n] * (1.f + dm * 0.5f);
  const float dt_tau = DT_ / tau_m[n];
  const float vth_n  = vth[n];
  const float rst_n  = rstv[n];
  const float dka = 0.99900049983337496f;  // exp(-DT/100)
  const float dks = 0.98019867330675525f;  // exp(-DT/5)

  float v[8], rf[8], is[8], ad[8];
  #pragma unroll
  for (int r = 0; r < 8; r++) {
    const int idx = (b0 + r + 8 * hi) * N_ + n;
    v[r] = v0[idx]; rf[r] = rf0[idx]; is[r] = is0[idx]; ad[r] = ad0[idx];
  }

  // W -> f16 WMMA B-operand registers, once. B[k][col] = W[n][k], K padded 80->96.
  BU Bw[3];
  #pragma unroll
  for (int c = 0; c < 3; c++) {
    #pragma unroll
    for (int r = 0; r < 8; r++) {
      const int k0 = c * 32 + (hi ? 16 : 0) + 2 * r;
      const float w0 = (k0     < N_) ? W[n * N_ + k0    ] : 0.f;
      const float w1 = (k0 + 1 < N_) ? W[n * N_ + k0 + 1] : 0.f;
      Bw[c].u[r] = pack_h2(w0, w1);
    }
  }

  // Software pipeline: noise for step 0 pre-loaded into registers.
  float nz[8];
  #pragma unroll
  for (int r = 0; r < 8; r++)
    nz[r] = noise[(size_t)(b0 + r + 8 * hi) * N_ + n];

  __syncthreads();  // one-time: LDS zero + W preload visible

  const int m = col;  // A-matrix row handled by this lane
  for (int t = 0; t < T_; t++) {
    _Float16* sb = spk + (t & 1) * (16 * STR);
    float*    vp = out_v + ((size_t)t * B_ + b0) * N_;

    #pragma unroll
    for (int r = 0; r < 8; r++) {
      const int  row = r + 8 * hi;
      const float it = is[r] + drv - ad[r] + nz[r] * 0.5f;
      const bool act = (rf[r] <= 0.f);
      v[r] += act ? dt_tau * (it - v[r]) : 0.f;
      const bool spike = act && (v[r] >= vth_n);
      const float s = spike ? 1.f : 0.f;
      v[r] = spike ? rst_n : v[r];
      vp[row * N_ + n] = v[r];
      rf[r] = fmaxf((spike ? 1.f : rf[r]) - DT_, 0.f);
      ad[r] = ad[r] * dka + 2.f * s;
      sb[row * STR + n] = (_Float16)s;
    }

    // Issue next step's noise loads now: latency overlaps barrier+WMMA+pointwise.
    float nz_next[8];
    if (t + 1 < T_) {
      const float* np1 = noise + ((size_t)(t + 1) * B_ + b0) * N_;
      #pragma unroll
      for (int r = 0; r < 8; r++) nz_next[r] = np1[(r + 8 * hi) * N_ + n];
      __builtin_prefetch(np1 + (size_t)B_ * N_ + m * N_ + n, 0, 0);  // t+2 -> L2
    } else {
      #pragma unroll
      for (int r = 0; r < 8; r++) nz_next[r] = 0.f;
    }

    lds_barrier();  // spikes visible; waits dscnt only

    // A (spikes, 16x96 f16) via b128 LDS loads in contiguous k order (k = 16*hi + s), same as Bw.
    AU A[3];
    #pragma unroll
    for (int c = 0; c < 3; c++) {
      const _Float16* p = sb + m * STR + c * 32 + (hi ? 16 : 0);
      A[c].q[0] = *(const uint4*)(p);
      A[c].q[1] = *(const uint4*)(p + 8);
    }

    v8f acc = {};
    acc = __builtin_amdgcn_wmma_f32_16x16x32_f16(false, A[0].h, false, Bw[0].h,
                                                 (short)0, acc, false, false);
    acc = __builtin_amdgcn_wmma_f32_16x16x32_f16(false, A[1].h, false, Bw[1].h,
                                                 (short)0, acc, false, false);
    acc = __builtin_amdgcn_wmma_f32_16x16x32_f16(false, A[2].h, false, Bw[2].h,
                                                 (short)0, acc, false, false);
    #pragma unroll
    for (int r = 0; r < 8; r++) is[r] = is[r] * dks + acc[r];

    // Motor output: wave 0, lanes 0-15 read the spike LDS tile.
    if (tid < 16) {
      float lf = 0.f, le = 0.f, rfm = 0.f, re = 0.f;
      #pragma unroll
      for (int k = 0; k < 5; k++) {
        lf  += (float)sb[tid * STR + 30 + k];
        le  += (float)sb[tid * STR + 35 + k];
        rfm += (float)sb[tid * STR + 70 + k];
        re  += (float)sb[tid * STR + 75 + k];
      }
      lf *= 0.2f; le *= 0.2f; rfm *= 0.2f; re *= 0.2f;
      float* mp = out_motor + ((size_t)t * B_ + b0 + tid) * 6;
      mp[0] = lf;  mp[1] = le;  mp[2] = le - lf;
      mp[3] = rfm; mp[4] = re;  mp[5] = re - rfm;
    }
    // No trailing barrier: next step writes the OTHER spike buffer; reuse of
    // this buffer at t+2 is fenced by the barrier at t+1 (readers drain their
    // ds ops before signaling it).

    #pragma unroll
    for (int r = 0; r < 8; r++) nz[r] = nz_next[r];
  }
}

extern "C" void kernel_launch(void* const* d_in, const int* in_sizes, int n_in,
                              void* d_out, int out_size, void* d_ws, size_t ws_size,
                              hipStream_t stream) {
  (void)in_sizes; (void)n_in; (void)out_size; (void)d_ws; (void)ws_size;
  const float* v0   = (const float*)d_in[0];
  const float* rf0  = (const float*)d_in[1];
  const float* is0  = (const float*)d_in[2];
  const float* ad0  = (const float*)d_in[3];
  const float* nz   = (const float*)d_in[4];
  const float* dm   = (const float*)d_in[5];
  const float* W    = (const float*)d_in[6];
  const float* tau  = (const float*)d_in[7];
  const float* vthr = (const float*)d_in[8];
  const float* drv  = (const float*)d_in[9];
  const float* rst  = (const float*)d_in[10];
  float* out_v = (float*)d_out;
  float* out_m = out_v + (size_t)T_ * B_ * N_;  // v_traj then motor_traj

  dim3 grid(B_ / 16), block(160);
  hipLaunchKernelGGL(SpikingCPG_kernel, grid, block, 0, stream,
                     v0, rf0, is0, ad0, nz, dm, W, tau, vthr, drv, rst,
                     out_v, out_m);
}
